// EvenLamerGAT_73504070303823
// MI455X (gfx1250) — matrix operand order repaired and checked by reading, not compiled, not run on hardware
//
#include <hip/hip_runtime.h>

#define NNODES 50000
#define NEDGES 800000
#define ETOT   (NEDGES + NNODES)

typedef __bf16 bf16t;
typedef __attribute__((ext_vector_type(16))) __bf16 v16bf;
typedef __attribute__((ext_vector_type(8)))  float  v8f;

// ---------------------------------------------------------------- utilities
__global__ void k_fill(float* __restrict__ p, float v, long n) {
  long i = (long)blockIdx.x * blockDim.x + threadIdx.x;
  long stride = (long)gridDim.x * blockDim.x;
  for (; i < n; i += stride) p[i] = v;
}

// W [K, Nc] f32 row-major  ->  WT [Nc, K] bf16 (column-major of W)
__global__ void k_w_to_bf16t(const float* __restrict__ W, bf16t* __restrict__ WT,
                             int K, int Nc) {
  int i = blockIdx.x * blockDim.x + threadIdx.x;
  if (i >= K * Nc) return;
  int k = i % K, n = i / K;
  WT[i] = (bf16t)W[(size_t)k * Nc + n];
}

__device__ __forceinline__ void atomicMaxF(float* addr, float val) {
  // IEEE ordering trick: works for mixed signs when initialized to -large
  if (val >= 0.0f) atomicMax((int*)addr, __float_as_int(val));
  else             atomicMin((unsigned int*)addr, __float_as_uint(val));
}

// ------------------------------------------------------- WMMA bf16 GEMM+bias
// out[M,Nc] = X[M,K] (f32 -> bf16) * W[K,Nc] + bias.  WT is [Nc,K] bf16.
// One wave computes a 16x16 tile; block = 128 threads = 4 waves = 64 N-cols.
// M=50000 is divisible by 16, K in {128,256} divisible by 32.
__global__ void k_gemm_wmma(const float* __restrict__ X, const bf16t* __restrict__ WT,
                            const float* __restrict__ bias, float* __restrict__ out,
                            int K, int Nc) {
  const int lane  = threadIdx.x & 31;
  const int wave  = threadIdx.x >> 5;
  const int r     = lane & 15;
  const int hi    = lane >> 4;                   // 0 or 1
  const int mbase = blockIdx.x << 4;
  const int nbase = (blockIdx.y << 6) + (wave << 4);
  // A fragment rows: lane holds row (mbase+r); 16 contiguous K at hi*16
  const float* __restrict__ arow = X + (size_t)(mbase + r) * K + (hi << 4);
  // B fragment cols: lane holds col (nbase+r); 16 contiguous K at hi*16
  const bf16t* __restrict__ brow = WT + (size_t)(nbase + r) * K + (hi << 4);
  v8f acc = {};
  for (int k0 = 0; k0 < K; k0 += 32) {
    v16bf a, b;
#pragma unroll
    for (int j = 0; j < 8; ++j) {
      a[j]     = (bf16t)arow[k0 + j];
      a[j + 8] = (bf16t)arow[k0 + 8 + j];
    }
#pragma unroll
    for (int j = 0; j < 16; ++j) b[j] = brow[k0 + j];
    acc = __builtin_amdgcn_wmma_f32_16x16x32_bf16(
        false, a, false, b, (short)0, acc, false, false);
  }
  const int   n  = nbase + r;
  const int   m0 = mbase + (hi << 3);
  const float bv = bias[n];
#pragma unroll
  for (int i = 0; i < 8; ++i)
    out[(size_t)(m0 + i) * Nc + n] = acc[i] + bv;
}

// ------------------------------------------------------------- edge kernels
// pass 1: e = sum_c leaky_relu(xl[src]+xr[dst]) * att ; segment max via atomics
__global__ void k_edge_scores(const float* __restrict__ xl, const float* __restrict__ xr,
                              const float* __restrict__ att,
                              const int* __restrict__ srcA, const int* __restrict__ dstA,
                              int H, int C, float* __restrict__ e,
                              float* __restrict__ emax) {
  long t = (long)blockIdx.x * blockDim.x + threadIdx.x;
  if (t >= (long)ETOT * H) return;
  int edge = (int)(t / H), h = (int)(t % H);
  int src, dst;
  if (edge < NEDGES) { src = srcA[edge]; dst = dstA[edge]; }
  else               { src = dst = edge - NEDGES; }
  const float* __restrict__ a = xl + (size_t)src * (H * C) + h * C;
  const float* __restrict__ b = xr + (size_t)dst * (H * C) + h * C;
  const float* __restrict__ w = att + h * C;
  float s = 0.f;
#pragma unroll 8
  for (int c = 0; c < C; ++c) {
    float m = a[c] + b[c];
    m = m > 0.f ? m : m * 0.2f;
    s += m * w[c];
  }
  e[t] = s;
  atomicMaxF(&emax[(size_t)dst * H + h], s);
}

// pass 2: ex = exp(e - emax[dst]); denom[dst] += ex (in-place overwrite of e)
__global__ void k_edge_softnorm(const int* __restrict__ dstA, int H,
                                float* __restrict__ e, const float* __restrict__ emax,
                                float* __restrict__ denom) {
  long t = (long)blockIdx.x * blockDim.x + threadIdx.x;
  if (t >= (long)ETOT * H) return;
  int edge = (int)(t / H), h = (int)(t % H);
  int dst = edge < NEDGES ? dstA[edge] : edge - NEDGES;
  float ex = __expf(e[t] - emax[(size_t)dst * H + h]);
  e[t] = ex;
  atomicAdd(&denom[(size_t)dst * H + h], ex);
}

// pass 3: out[dst] += (ex/denom[dst]) * xl[src]; one thread per (edge, 4 channels)
__global__ void k_edge_scatter(const float* __restrict__ xl,
                               const int* __restrict__ srcA, const int* __restrict__ dstA,
                               int H, int C, const float* __restrict__ ex,
                               const float* __restrict__ denom, float* __restrict__ acc) {
  const int D  = H * C;
  const int D4 = D >> 2;
  long t = (long)blockIdx.x * blockDim.x + threadIdx.x;
  if (t >= (long)ETOT * D4) return;
  int edge = (int)(t / D4);
  int q    = ((int)(t % D4)) << 2;     // channel offset (4-aligned, within a head)
  int h    = q / C;
  int src, dst;
  if (edge < NEDGES) { src = srcA[edge]; dst = dstA[edge]; }
  else               { src = dst = edge - NEDGES; }
  float alpha = ex[(size_t)edge * H + h] / denom[(size_t)dst * H + h];
  const float4 v = *(const float4*)(xl + (size_t)src * D + q);
  float* o = acc + (size_t)dst * D + q;
  atomicAdd(o + 0, v.x * alpha);
  atomicAdd(o + 1, v.y * alpha);
  atomicAdd(o + 2, v.z * alpha);
  atomicAdd(o + 3, v.w * alpha);
}

__global__ void k_bias_elu(float* __restrict__ h, const float* __restrict__ bias,
                           long n, int D) {
  long t = (long)blockIdx.x * blockDim.x + threadIdx.x;
  if (t >= n) return;
  float v = h[t] + bias[t % D];
  h[t] = v > 0.f ? v : (__expf(v) - 1.f);
}

// bias add + log_softmax over 64 channels; one wave32 per node (2 chans/lane)
__global__ void k_finalize(const float* __restrict__ h2, const float* __restrict__ bias,
                           float* __restrict__ out) {
  int wave = threadIdx.x >> 5, lane = threadIdx.x & 31;
  int node = blockIdx.x * 8 + wave;
  if (node >= NNODES) return;
  float v0 = h2[(size_t)node * 64 + lane]      + bias[lane];
  float v1 = h2[(size_t)node * 64 + 32 + lane] + bias[32 + lane];
  out[(size_t)node * 64 + lane]      = v0;
  out[(size_t)node * 64 + 32 + lane] = v1;
  float mx = fmaxf(v0, v1);
#pragma unroll
  for (int off = 16; off; off >>= 1) mx = fmaxf(mx, __shfl_xor(mx, off, 32));
  float s = __expf(v0 - mx) + __expf(v1 - mx);
#pragma unroll
  for (int off = 16; off; off >>= 1) s += __shfl_xor(s, off, 32);
  float lse = mx + __logf(s);
  float* ls = out + (size_t)NNODES * 64;
  ls[(size_t)node * 64 + lane]      = v0 - lse;
  ls[(size_t)node * 64 + 32 + lane] = v1 - lse;
}

// ------------------------------------------------------------------ launch
extern "C" void kernel_launch(void* const* d_in, const int* in_sizes, int n_in,
                              void* d_out, int out_size, void* d_ws, size_t ws_size,
                              hipStream_t stream) {
  const float* x     = (const float*)d_in[0];
  const int*   ei    = (const int*)d_in[1];        // [2, E] (JAX default int32)
  const float* Wl1   = (const float*)d_in[2];
  const float* bl1   = (const float*)d_in[3];
  const float* Wr1   = (const float*)d_in[4];
  const float* br1   = (const float*)d_in[5];
  const float* att1  = (const float*)d_in[6];
  const float* bias1 = (const float*)d_in[7];
  const float* Wl2   = (const float*)d_in[8];
  const float* bl2   = (const float*)d_in[9];
  const float* Wr2   = (const float*)d_in[10];
  const float* br2   = (const float*)d_in[11];
  const float* att2  = (const float*)d_in[12];
  const float* bias2 = (const float*)d_in[13];
  float* outp = (float*)d_out;
  const int* srcA = ei;
  const int* dstA = ei + NEDGES;

  // workspace layout (f32 counts)
  float* w = (float*)d_ws;
  size_t off = 0;
  float* xl1   = w + off; off += (size_t)NNODES * 256;
  float* xr1   = w + off; off += (size_t)NNODES * 256;
  float* hbuf  = w + off; off += (size_t)NNODES * 256;
  float* xl2   = w + off; off += (size_t)NNODES * 64;
  float* xr2   = w + off; off += (size_t)NNODES * 64;
  float* e1    = w + off; off += (size_t)ETOT * 8;
  float* emax1 = w + off; off += (size_t)NNODES * 8;
  float* den1  = w + off; off += (size_t)NNODES * 8;
  float* e2    = w + off; off += (size_t)ETOT;
  float* emax2 = w + off; off += (size_t)NNODES;
  float* den2  = w + off; off += (size_t)NNODES;
  float* h2    = w + off; off += (size_t)NNODES * 64;
  bf16t* wl1t = (bf16t*)(w + off);
  bf16t* wr1t = wl1t + 128 * 256;
  bf16t* wl2t = wr1t + 128 * 256;
  bf16t* wr2t = wl2t + 256 * 64;

  auto blocks = [](long n, int b) { return (unsigned)((n + b - 1) / b); };

  // weight transpose -> bf16
  k_w_to_bf16t<<<blocks(128 * 256, 256), 256, 0, stream>>>(Wl1, wl1t, 128, 256);
  k_w_to_bf16t<<<blocks(128 * 256, 256), 256, 0, stream>>>(Wr1, wr1t, 128, 256);
  k_w_to_bf16t<<<blocks(256 * 64, 256), 256, 0, stream>>>(Wl2, wl2t, 256, 64);
  k_w_to_bf16t<<<blocks(256 * 64, 256), 256, 0, stream>>>(Wr2, wr2t, 256, 64);

  // init accumulators
  k_fill<<<2048, 256, 0, stream>>>(emax1, -3.0e38f, (long)NNODES * 8);
  k_fill<<<2048, 256, 0, stream>>>(den1, 0.0f, (long)NNODES * 8);
  k_fill<<<4096, 256, 0, stream>>>(hbuf, 0.0f, (long)NNODES * 256);
  k_fill<<<1024, 256, 0, stream>>>(emax2, -3.0e38f, (long)NNODES);
  k_fill<<<1024, 256, 0, stream>>>(den2, 0.0f, (long)NNODES);
  k_fill<<<2048, 256, 0, stream>>>(h2, 0.0f, (long)NNODES * 64);

  // ---- layer 1 ----
  dim3 g1(NNODES / 16, 256 / 64);
  k_gemm_wmma<<<g1, 128, 0, stream>>>(x, wl1t, bl1, xl1, 128, 256);
  k_gemm_wmma<<<g1, 128, 0, stream>>>(x, wr1t, br1, xr1, 128, 256);

  k_edge_scores<<<blocks((long)ETOT * 8, 256), 256, 0, stream>>>(
      xl1, xr1, att1, srcA, dstA, 8, 32, e1, emax1);
  k_edge_softnorm<<<blocks((long)ETOT * 8, 256), 256, 0, stream>>>(
      dstA, 8, e1, emax1, den1);
  k_edge_scatter<<<blocks((long)ETOT * 64, 256), 256, 0, stream>>>(
      xl1, srcA, dstA, 8, 32, e1, den1, hbuf);
  k_bias_elu<<<blocks((long)NNODES * 256, 256), 256, 0, stream>>>(
      hbuf, bias1, (long)NNODES * 256, 256);

  // ---- layer 2 ----
  dim3 g2(NNODES / 16, 1);
  k_gemm_wmma<<<g2, 128, 0, stream>>>(hbuf, wl2t, bl2, xl2, 256, 64);
  k_gemm_wmma<<<g2, 128, 0, stream>>>(hbuf, wr2t, br2, xr2, 256, 64);

  k_edge_scores<<<blocks((long)ETOT, 256), 256, 0, stream>>>(
      xl2, xr2, att2, srcA, dstA, 1, 64, e2, emax2);
  k_edge_softnorm<<<blocks((long)ETOT, 256), 256, 0, stream>>>(
      dstA, 1, e2, emax2, den2);
  k_edge_scatter<<<blocks((long)ETOT * 16, 256), 256, 0, stream>>>(
      xl2, srcA, dstA, 1, 64, e2, den2, h2);

  k_finalize<<<blocks((long)NNODES, 8), 256, 0, stream>>>(h2, bias2, outp);
}
